// MCSwinTransformerBlock_63084479643870
// MI455X (gfx1250) — compile-verified, matrix operand order checked by reading, not run on hardware
//
#include <hip/hip_runtime.h>
#include <hip/hip_bf16.h>
#include <math.h>

// Shapes: B=32, N=4096, C=256, WS=64, SHIFT=32, H=8, HD=32, NW=64, Bw=2048, HID=1024
// M_tok = B*N = 131072 rows everywhere.

typedef __attribute__((ext_vector_type(16))) _Float16 v16h;
typedef __attribute__((ext_vector_type(8)))  _Float16 v8h;
typedef __attribute__((ext_vector_type(8)))  float    v8f;

// ---------------------------------------------------------------------------
// WMMA helpers (gfx1250: V_WMMA_F32_16X16X32_F16, wave32)
// A-frag: lane L holds row M=L%16; halves i=0..7 -> K = i + 8*(L/16),
//         i=8..15 -> K = 16 + (i-8) + 8*(L/16).  => two contiguous 16B loads.
// B-frag: identical layout with N=L%16 over rows of W[N][K] (weights are [out,in]).
// C/D:    vgpr i, lane L: M = i + 8*(L/16), N = L%16.
// ---------------------------------------------------------------------------
__device__ __forceinline__ v16h load_frag(const _Float16* __restrict__ p, int ld) {
  const int lane = threadIdx.x & 31;
  const int r    = lane & 15;
  const int hoff = (lane >> 4) << 3;     // 0 or 8
  const _Float16* q = p + r * ld + hoff;
  v8h lo = *(const v8h*)(q);
  v8h hi = *(const v8h*)(q + 16);
  v16h out;
#pragma unroll
  for (int i = 0; i < 8; ++i) { out[i] = lo[i]; out[i + 8] = hi[i]; }
  return out;
}

__device__ __forceinline__ v8f wmma16(v16h a, v16h b, v8f c) {
  return __builtin_amdgcn_wmma_f32_16x16x32_f16(false, a, false, b, (short)0, c,
                                                false, false);
}

// ---------------------------------------------------------------------------
// Weight conversion f32 -> f16
// ---------------------------------------------------------------------------
__global__ __launch_bounds__(256) void cvt_f16(const float* __restrict__ s,
                                               _Float16* __restrict__ d, int n) {
  int i = blockIdx.x * 256 + threadIdx.x;
  if (i < n) d[i] = (_Float16)s[i];
}

// pos_w [C_out, C_in, 3] -> taps[k][co*256+ci]
__global__ __launch_bounds__(256) void cvt_posw(const float* __restrict__ s,
                                                _Float16* __restrict__ d, int n) {
  int i = blockIdx.x * 256 + threadIdx.x;
  if (i < n) {
    int k = i % 3;
    int o = i / 3;
    d[k * 65536 + o] = (_Float16)s[i];
  }
}

// ---------------------------------------------------------------------------
// LayerNorm over C=256, one wave per row; optional cyclic shift fold
// dst row m pulls src row (m & ~4095) | ((m+32) & 4095)  (roll -SHIFT)
// ---------------------------------------------------------------------------
__global__ __launch_bounds__(256) void ln_kernel(const float* __restrict__ src,
                                                 const float* __restrict__ w,
                                                 const float* __restrict__ b,
                                                 _Float16* __restrict__ dst,
                                                 int shifted) {
  const int wv   = threadIdx.x >> 5;
  const int lane = threadIdx.x & 31;
  const int m    = blockIdx.x * 8 + wv;
  int srow = m;
  if (shifted) srow = (m & ~4095) | ((m + 32) & 4095);
  const float* xr = src + (size_t)srow * 256;
  float vb[8];
  float s = 0.f;
#pragma unroll
  for (int j = 0; j < 8; ++j) { vb[j] = xr[lane * 8 + j]; s += vb[j]; }
#pragma unroll
  for (int off = 1; off < 32; off <<= 1) s += __shfl_xor(s, off, 32);
  const float mean = s * (1.0f / 256.0f);
  float s2 = 0.f;
#pragma unroll
  for (int j = 0; j < 8; ++j) { float d = vb[j] - mean; s2 += d * d; }
#pragma unroll
  for (int off = 1; off < 32; off <<= 1) s2 += __shfl_xor(s2, off, 32);
  const float rs = rsqrtf(s2 * (1.0f / 256.0f) + 1e-5f);
  _Float16* dr = dst + (size_t)m * 256;
#pragma unroll
  for (int j = 0; j < 8; ++j) {
    int c = lane * 8 + j;
    dr[c] = (_Float16)((vb[j] - mean) * rs * w[c] + b[c]);
  }
}

// ---------------------------------------------------------------------------
// Generic WMMA GEMM: out[M,N] = A[M,K](f16) * W[N,K]^T(f16) + epilogue
// Block tile 128x64, 8 waves of 32x32, BK=32, double-buffered LDS
// (stride 40 halves = 80B to avoid bank conflicts), 1 barrier per K-step.
// EPI: 0=qkv(bias + q-scale, f16 ldN=768)  1=proj(bias + roll+residual, f32)
//      2=fc1(bias + exact GELU, f16 ldN=1024)  3=fc2(bias + residual, f32)
// ---------------------------------------------------------------------------
template <int EPI>
__global__ __launch_bounds__(256) void gemm_kernel(
    const _Float16* __restrict__ A, const _Float16* __restrict__ W,
    const float* __restrict__ bias, const float* __restrict__ resid,
    _Float16* __restrict__ outh, float* __restrict__ outf, int K) {
  __shared__ _Float16 As[2][128 * 40];
  __shared__ _Float16 Bs[2][64 * 40];
  const int lane = threadIdx.x & 31;
  const int wv   = threadIdx.x >> 5;
  const int wm   = wv >> 1;
  const int wn   = wv & 1;
  const int m0   = blockIdx.y * 128;
  const int n0   = blockIdx.x * 64;
  v8f zf = {0.f, 0.f, 0.f, 0.f, 0.f, 0.f, 0.f, 0.f};
  v8f acc[2][2];
  acc[0][0] = zf; acc[0][1] = zf; acc[1][0] = zf; acc[1][1] = zf;

  const int ar = threadIdx.x >> 2;                 // A copy: row, 2 chunks/thread
  const int ac = (threadIdx.x & 3) << 3;
  const int br = threadIdx.x >> 2;                 // B copy: 1 chunk/thread
  const int bc = (threadIdx.x & 3) << 3;

  auto stage = [&](int k0, int buf) {
#pragma unroll
    for (int i = 0; i < 2; ++i) {
      int r = ar + i * 64;
      *(uint4*)&As[buf][r * 40 + ac] =
          *(const uint4*)&A[(size_t)(m0 + r) * K + k0 + ac];
    }
    *(uint4*)&Bs[buf][br * 40 + bc] =
        *(const uint4*)&W[(size_t)(n0 + br) * K + k0 + bc];
  };

  stage(0, 0);
  __syncthreads();
  const int nk = K >> 5;
  for (int kt = 0; kt < nk; ++kt) {
    const int cur = kt & 1;
    if (kt + 1 < nk) stage((kt + 1) << 5, cur ^ 1);
    v16h a0 = load_frag(&As[cur][(wm * 32 + 0) * 40], 40);
    v16h a1 = load_frag(&As[cur][(wm * 32 + 16) * 40], 40);
    v16h b0 = load_frag(&Bs[cur][(wn * 32 + 0) * 40], 40);
    v16h b1 = load_frag(&Bs[cur][(wn * 32 + 16) * 40], 40);
    acc[0][0] = wmma16(a0, b0, acc[0][0]);
    acc[0][1] = wmma16(a0, b1, acc[0][1]);
    acc[1][0] = wmma16(a1, b0, acc[1][0]);
    acc[1][1] = wmma16(a1, b1, acc[1][1]);
    __syncthreads();
  }

  const int cn = lane & 15, ch = lane >> 4;
#pragma unroll
  for (int mi = 0; mi < 2; ++mi)
#pragma unroll
    for (int ni = 0; ni < 2; ++ni)
#pragma unroll
      for (int i = 0; i < 8; ++i) {
        const int gm = m0 + wm * 32 + mi * 16 + ch * 8 + i;
        const int gn = n0 + wn * 32 + ni * 16 + cn;
        float v = acc[mi][ni][i] + bias[gn];
        if (EPI == 0) {                          // qkv: scale q columns
          if (gn < 256) v *= 0.17677669529663687f;  // HD^-0.5
          outh[(size_t)gm * 768 + gn] = (_Float16)v;
        } else if (EPI == 1) {                   // proj: roll +SHIFT, residual
          int n2 = (gm + 32) & 4095;
          size_t di = ((size_t)((gm & ~4095) | n2)) * 256 + gn;
          outf[di] = resid[di] + v;
        } else if (EPI == 2) {                   // fc1: exact GELU
          float g = 0.5f * v * (1.0f + erff(v * 0.70710678118654752f));
          outh[(size_t)gm * 1024 + gn] = (_Float16)g;
        } else {                                 // fc2: residual, f32 out
          size_t di = (size_t)gm * 256 + gn;
          outf[di] = resid[di] + v;
        }
      }
}

// ---------------------------------------------------------------------------
// Windowed attention, one block (4 waves) per (window bw, head hh).
// S = q k^T (q pre-scaled), + mask[bw%64], softmax, O = P v  -> y f16
// V is staged with b128 loads into LDS (ps area), then LDS-transposed into vt.
// Canonical Swin shift mask is nonzero only for the last window (nw==63);
// the branch is block-uniform so EXEC is all-ones around every WMMA.
// ---------------------------------------------------------------------------
__global__ __launch_bounds__(128) void attn_kernel(const _Float16* __restrict__ qkv,
                                                   const float* __restrict__ mask,
                                                   _Float16* __restrict__ y) {
  __shared__ _Float16 qs[64 * 32];
  __shared__ _Float16 ks[64 * 32];
  __shared__ _Float16 vt[32 * 64];   // V transposed: vt[d][t]
  __shared__ _Float16 ps[64 * 64];   // V staging first, then softmax probs
  const int bw   = blockIdx.x >> 3;
  const int hh   = blockIdx.x & 7;
  const int tid  = threadIdx.x;
  const int lane = tid & 31;
  const int w    = tid >> 5;

#pragma unroll
  for (int i = 0; i < 2; ++i) {
    int id = tid + i * 128;                 // 0..255 : 64 rows x 4 chunks
    int t = id >> 2, cc = (id & 3) << 3;
    const _Float16* base = qkv + (size_t)(bw * 64 + t) * 768 + hh * 32 + cc;
    *(uint4*)&qs[t * 32 + cc] = *(const uint4*)(base);
    *(uint4*)&ks[t * 32 + cc] = *(const uint4*)(base + 256);
    *(uint4*)&ps[t * 32 + cc] = *(const uint4*)(base + 512);   // V row-major
  }
  __syncthreads();
#pragma unroll
  for (int i = 0; i < 16; ++i) {            // LDS -> LDS transpose of V
    int id = tid + i * 128;                 // 0..2047
    int t = id >> 5, d = id & 31;
    vt[d * 64 + t] = ps[t * 32 + d];
  }
  __syncthreads();

  v8f zf = {0.f, 0.f, 0.f, 0.f, 0.f, 0.f, 0.f, 0.f};
  v16h aq = load_frag(&qs[(w * 16) * 32], 32);
  v8f s[4];
#pragma unroll
  for (int n = 0; n < 4; ++n) {
    v16h bk = load_frag(&ks[(n * 16) * 32], 32);
    s[n] = wmma16(aq, bk, zf);
  }

  const int cn = lane & 15, ch = lane >> 4;
  if ((bw & 63) == 63) {                    // only last window has nonzero mask
    const float* mrow = mask + (size_t)63 * 4096;
#pragma unroll
    for (int n = 0; n < 4; ++n)
#pragma unroll
      for (int i = 0; i < 8; ++i) {
        int row = w * 16 + ch * 8 + i;
        s[n][i] += mrow[row * 64 + n * 16 + cn];
      }
  }

#pragma unroll
  for (int i = 0; i < 8; ++i) {
    float mx = fmaxf(fmaxf(s[0][i], s[1][i]), fmaxf(s[2][i], s[3][i]));
#pragma unroll
    for (int off = 1; off < 16; off <<= 1) mx = fmaxf(mx, __shfl_xor(mx, off, 32));
    float e0 = __expf(s[0][i] - mx);
    float e1 = __expf(s[1][i] - mx);
    float e2 = __expf(s[2][i] - mx);
    float e3 = __expf(s[3][i] - mx);
    float tot = e0 + e1 + e2 + e3;
#pragma unroll
    for (int off = 1; off < 16; off <<= 1) tot += __shfl_xor(tot, off, 32);
    float inv = 1.0f / tot;
    int row = w * 16 + ch * 8 + i;
    ps[row * 64 + 0 * 16 + cn] = (_Float16)(e0 * inv);
    ps[row * 64 + 1 * 16 + cn] = (_Float16)(e1 * inv);
    ps[row * 64 + 2 * 16 + cn] = (_Float16)(e2 * inv);
    ps[row * 64 + 3 * 16 + cn] = (_Float16)(e3 * inv);
  }
  __syncthreads();

#pragma unroll
  for (int nt = 0; nt < 2; ++nt) {
    v8f o = zf;
#pragma unroll
    for (int kk = 0; kk < 2; ++kk) {
      v16h ap = load_frag(&ps[(w * 16) * 64 + kk * 32], 64);
      v16h bv = load_frag(&vt[(nt * 16) * 64 + kk * 32], 64);
      o = wmma16(ap, bv, o);
    }
#pragma unroll
    for (int i = 0; i < 8; ++i) {
      int row = w * 16 + ch * 8 + i;
      y[(size_t)(bw * 64 + row) * 256 + hh * 32 + nt * 16 + cn] = (_Float16)o[i];
    }
  }
}

// ---------------------------------------------------------------------------
// LePE: full Conv1d over window axis (C out, C in, kernel 3, pad 1 per window)
// One block per window. y += lepe + pos_b. 3 tap-GEMMs vs padded V in LDS.
// ---------------------------------------------------------------------------
__global__ __launch_bounds__(256) void lepe_kernel(const _Float16* __restrict__ qkv,
                                                   const _Float16* __restrict__ taps,
                                                   const float* __restrict__ pos_b,
                                                   _Float16* __restrict__ y) {
  __shared__ _Float16 vp[66 * 256];          // rows 0 and 65 are zero padding
  const int bw   = blockIdx.x;
  const int tid  = threadIdx.x;
  const int lane = tid & 31;
  const int wv   = tid >> 5;                 // 0..7, owns N cols [wv*32, wv*32+32)

  vp[tid] = (_Float16)0.f;
  vp[65 * 256 + tid] = (_Float16)0.f;
#pragma unroll
  for (int i = 0; i < 8; ++i) {
    int id = tid + i * 256;                  // 0..2047 : 64 rows x 32 chunks
    int t = id >> 5, cc = (id & 31) << 3;
    *(uint4*)&vp[(t + 1) * 256 + cc] =
        *(const uint4*)&qkv[(size_t)(bw * 64 + t) * 768 + 512 + cc];
  }
  __syncthreads();

  v8f zf = {0.f, 0.f, 0.f, 0.f, 0.f, 0.f, 0.f, 0.f};
  v8f acc[4][2];
#pragma unroll
  for (int mt = 0; mt < 4; ++mt) { acc[mt][0] = zf; acc[mt][1] = zf; }

  for (int k = 0; k < 3; ++k) {
    const _Float16* Wk = taps + k * 65536;
    for (int kc = 0; kc < 8; ++kc) {
      v16h b0 = load_frag(&Wk[(size_t)(wv * 32 + 0) * 256 + kc * 32], 256);
      v16h b1 = load_frag(&Wk[(size_t)(wv * 32 + 16) * 256 + kc * 32], 256);
#pragma unroll
      for (int mt = 0; mt < 4; ++mt) {
        v16h a = load_frag(&vp[(mt * 16 + k) * 256 + kc * 32], 256);
        acc[mt][0] = wmma16(a, b0, acc[mt][0]);
        acc[mt][1] = wmma16(a, b1, acc[mt][1]);
      }
    }
  }

  const int cn = lane & 15, ch = lane >> 4;
#pragma unroll
  for (int mt = 0; mt < 4; ++mt)
#pragma unroll
    for (int nt = 0; nt < 2; ++nt)
#pragma unroll
      for (int i = 0; i < 8; ++i) {
        int row = mt * 16 + ch * 8 + i;
        int col = wv * 32 + nt * 16 + cn;
        size_t di = (size_t)(bw * 64 + row) * 256 + col;
        float cur = (float)y[di];
        y[di] = (_Float16)(cur + acc[mt][nt][i] + pos_b[col]);
      }
}

// ---------------------------------------------------------------------------
// Orchestration
// ---------------------------------------------------------------------------
extern "C" void kernel_launch(void* const* d_in, const int* in_sizes, int n_in,
                              void* d_out, int out_size, void* d_ws, size_t ws_size,
                              hipStream_t stream) {
  const float* x     = (const float*)d_in[0];
  const float* amask = (const float*)d_in[1];
  const float* n1w   = (const float*)d_in[2];
  const float* n1b   = (const float*)d_in[3];
  const float* qkvw  = (const float*)d_in[4];
  const float* qkvb  = (const float*)d_in[5];
  const float* posw  = (const float*)d_in[6];
  const float* posb  = (const float*)d_in[7];
  const float* projw = (const float*)d_in[8];
  const float* projb = (const float*)d_in[9];
  const float* n2w   = (const float*)d_in[10];
  const float* n2b   = (const float*)d_in[11];
  const float* fc1w  = (const float*)d_in[12];
  const float* fc1b  = (const float*)d_in[13];
  const float* fc2w  = (const float*)d_in[14];
  const float* fc2b  = (const float*)d_in[15];

  char* ws = (char*)d_ws;
  _Float16* hsw  = (_Float16*)(ws);                    // 64MB  (LN1 out; later y)
  _Float16* qkvB = (_Float16*)(ws + 67108864ull);      // 192MB (qkv; later h2)
  float*    x1   = (float*)   (ws + 268435456ull);     // 128MB
  _Float16* m1   = (_Float16*)(ws + 402653184ull);     // 256MB (fc1 out)
  char* wb = ws + 671088640ull;                        // ~2MB f16 weights
  _Float16* qkvW16  = (_Float16*)(wb);                 // 768*256
  _Float16* taps    = (_Float16*)(wb + 393216);        // 3 * 256*256
  _Float16* projW16 = (_Float16*)(wb + 786432);        // 256*256
  _Float16* fc1W16  = (_Float16*)(wb + 917504);        // 1024*256
  _Float16* fc2W16  = (_Float16*)(wb + 1441792);       // 256*1024
  _Float16* y  = hsw;   // alias: hsw dead after QKV GEMM
  _Float16* h2 = qkvB;  // alias: qkv dead after LePE

  // f16 weight staging
  cvt_f16 <<<(196608 + 255) / 256, 256, 0, stream>>>(qkvw, qkvW16, 196608);
  cvt_posw<<<(196608 + 255) / 256, 256, 0, stream>>>(posw, taps, 196608);
  cvt_f16 <<<(65536  + 255) / 256, 256, 0, stream>>>(projw, projW16, 65536);
  cvt_f16 <<<(262144 + 255) / 256, 256, 0, stream>>>(fc1w, fc1W16, 262144);
  cvt_f16 <<<(262144 + 255) / 256, 256, 0, stream>>>(fc2w, fc2W16, 262144);

  // LN1 with roll(-32) folded into source-row indexing -> f16 window layout
  ln_kernel<<<16384, 256, 0, stream>>>(x, n1w, n1b, hsw, 1);
  // QKV GEMM (q scaled in epilogue): [131072,256] x [768,256]^T
  gemm_kernel<0><<<dim3(12, 1024), 256, 0, stream>>>(hsw, qkvW16, qkvb, nullptr,
                                                     qkvB, nullptr, 256);
  // Windowed attention per (window, head)
  attn_kernel<<<16384, 128, 0, stream>>>(qkvB, amask, y);
  // LePE conv accumulated into y
  lepe_kernel<<<2048, 256, 0, stream>>>(qkvB, taps, posb, y);
  // proj GEMM + window reverse + roll(+32) + residual -> x1 (f32)
  gemm_kernel<1><<<dim3(4, 1024), 256, 0, stream>>>(y, projW16, projb, x,
                                                    nullptr, x1, 256);
  // LN2 -> h2 (f16)
  ln_kernel<<<16384, 256, 0, stream>>>(x1, n2w, n2b, h2, 0);
  // FC1 + exact GELU -> m1 (f16)
  gemm_kernel<2><<<dim3(16, 1024), 256, 0, stream>>>(h2, fc1W16, fc1b, nullptr,
                                                     m1, nullptr, 256);
  // FC2 + residual -> d_out (f32)
  gemm_kernel<3><<<dim3(4, 1024), 256, 0, stream>>>(m1, fc2W16, fc2b, x1,
                                                    nullptr, (float*)d_out, 1024);
}
